// HFGLMSelfAttentionFA_16028817948887
// MI455X (gfx1250) — matrix operand order repaired and checked by reading, not compiled, not run on hardware
//
#include <hip/hip_runtime.h>
#include <hip/hip_bf16.h>

// ---------------------------------------------------------------------------
// HFGLM self-attention block for MI455X (gfx1250), bf16 WMMA pipeline.
//   B=1, S=2048, H=2048, NH=16, HS=128
// Pipeline:
//   0) cast hidden fp32 -> bf16              (Xb   [S][H])
//   1) transpose+cast W_qkv -> bf16 N-major  (Wqt  [6144][2048])
//      transpose+cast W_dense -> bf16        (Wdt  [2048][2048])
//   2) WMMA GEMM: qkv = Xb @ W_qkv + b, split-store Q[h][s][d], K[h][s][d],
//      Vt[h][d][s] as bf16                   (wave tile 64x64, 16 WMMA/kstep)
//   3) per-wave flash attention (causal), ctx bf16 [S][H]
//   4) WMMA GEMM: out = ctx @ W_dense + b (fp32)
// Workspace: 72 MB.
// ---------------------------------------------------------------------------

typedef __bf16 bf16_t;
typedef __attribute__((ext_vector_type(16))) __bf16 v16bf;
typedef __attribute__((ext_vector_type(8)))  __bf16 v8bf;
typedef __attribute__((ext_vector_type(8)))  float  v8f;

#define S_LEN 2048
#define HID   2048
#define NQKV  6144
#define NHEAD 16
#define HSZ   128

static __device__ __forceinline__ v8f vzero8() {
  v8f z;
#pragma unroll
  for (int e = 0; e < 8; ++e) z[e] = 0.0f;
  return z;
}

static __device__ __forceinline__ v16bf cat16(v8bf lo, v8bf hi) {
  return __builtin_shufflevector(lo, hi, 0, 1, 2, 3, 4, 5, 6, 7,
                                          8, 9, 10, 11, 12, 13, 14, 15);
}

static __device__ __forceinline__ v8f wmma_bf16(v16bf a, v16bf b, v8f c) {
  // D = A(16x32) * B(32x16) + C, fp32 accum
  return __builtin_amdgcn_wmma_f32_16x16x32_bf16(false, a, false, b,
                                                 (short)0, c, false, false);
}

// A-fragment: lane r=lane&15 holds row M=r; contiguous k0+16b .. +15.
static __device__ __forceinline__ v16bf load_a_frag(const bf16_t* p) {
  v8bf lo = *(const v8bf*)p;         // k0 + 16b .. +7
  v8bf hi = *(const v8bf*)(p + 8);   // k0 + 16b + 8 .. +15
  return cat16(lo, hi);
}

// ---------------------------------------------------------------------------
// elementwise fp32 -> bf16 cast
// ---------------------------------------------------------------------------
__global__ __launch_bounds__(256) void k_cast_bf16(const float* __restrict__ x,
                                                   bf16_t* __restrict__ y, int n) {
  int i = blockIdx.x * 256 + threadIdx.x;
  if (i < n) y[i] = (bf16_t)x[i];
}

// ---------------------------------------------------------------------------
// W [K][N] fp32 row-major  ->  Wt [N][K] bf16 (N-major for B-fragments)
// ---------------------------------------------------------------------------
__global__ __launch_bounds__(256) void k_transpose_cast(const float* __restrict__ W,
                                                        bf16_t* __restrict__ Wt,
                                                        int K, int N) {
  __shared__ float tile[32][33];
  const int k0 = blockIdx.y * 32, n0 = blockIdx.x * 32;
  const int tx = threadIdx.x, ty = threadIdx.y;  // block (32,8)
#pragma unroll
  for (int rr = 0; rr < 4; ++rr)
    tile[ty + 8 * rr][tx] = W[(size_t)(k0 + ty + 8 * rr) * N + n0 + tx];
  __syncthreads();
#pragma unroll
  for (int rr = 0; rr < 4; ++rr)
    Wt[(size_t)(n0 + ty + 8 * rr) * K + k0 + tx] = (bf16_t)tile[tx][ty + 8 * rr];
}

// ---------------------------------------------------------------------------
// GEMM1: qkv = Xb[2048][2048] @ Wqt^T + b_qkv, split-store Q/K/Vt (bf16).
// Block = 256 threads = 8 waves; WG tile 128(M) x 256(N); wave tile 64x64.
// Per k-step: 4 A-frags + 4 B-frags -> 16 WMMAs (~32 FLOP/byte of frag load).
// ---------------------------------------------------------------------------
__global__ __launch_bounds__(256) void k_qkv_gemm(const bf16_t* __restrict__ X,
                                                  const bf16_t* __restrict__ Wt,
                                                  const float* __restrict__ bias,
                                                  bf16_t* __restrict__ Qb,
                                                  bf16_t* __restrict__ Kb,
                                                  bf16_t* __restrict__ Vt) {
  const int lane = threadIdx.x & 31;
  const int w    = threadIdx.x >> 5;
  const int r    = lane & 15;
  const int bsel = lane >> 4;
  const int m_base = blockIdx.y * 128 + (w & 1) * 64;
  const int n_base = blockIdx.x * 256 + (w >> 1) * 64;

  v8f acc[4][4];
#pragma unroll
  for (int i = 0; i < 4; ++i)
#pragma unroll
    for (int j = 0; j < 4; ++j) acc[i][j] = vzero8();

  for (int k0 = 0; k0 < HID; k0 += 32) {
    v16bf af[4];
#pragma unroll
    for (int i = 0; i < 4; ++i)
      af[i] = load_a_frag(X + (size_t)(m_base + 16 * i + r) * HID + k0 + 16 * bsel);
    v16bf bfr[4];
#pragma unroll
    for (int j = 0; j < 4; ++j)
      bfr[j] = *(const v16bf*)(Wt + (size_t)(n_base + 16 * j + r) * HID + k0 + 16 * bsel);
#pragma unroll
    for (int i = 0; i < 4; ++i)
#pragma unroll
      for (int j = 0; j < 4; ++j) acc[i][j] = wmma_bf16(af[i], bfr[j], acc[i][j]);
  }

#pragma unroll
  for (int j = 0; j < 4; ++j) {
    const int n    = n_base + 16 * j + r;
    const float bv = bias[n];
    const int sec  = n >> 11;       // 0=Q 1=K 2=V
    const int nn   = n & 2047;
    const int head = nn >> 7;
    const int d    = nn & 127;
#pragma unroll
    for (int i = 0; i < 4; ++i) {
#pragma unroll
      for (int v = 0; v < 8; ++v) {
        const int row  = m_base + 16 * i + v + 8 * bsel;  // C rows: v + 8b
        const bf16_t bb = (bf16_t)(acc[i][j][v] + bv);
        if (sec == 0)       Qb[((size_t)head * S_LEN + row) * HSZ + d] = bb;
        else if (sec == 1)  Kb[((size_t)head * S_LEN + row) * HSZ + d] = bb;
        else                Vt[((size_t)head * HSZ + d) * S_LEN + row] = bb;
      }
    }
  }
}

// ---------------------------------------------------------------------------
// Flash attention: 128 threads = 4 independent waves; each wave owns 32 query
// rows of one head, streams causal key blocks of 32, online softmax.
// Score C-frags -> LDS -> per-row softmax (lane=row) -> P A-frags from LDS.
// Intra-wave LDS ordering is guaranteed by the ISA (DS ops in-order per wave).
// ---------------------------------------------------------------------------
__global__ __launch_bounds__(128) void k_attn(const bf16_t* __restrict__ Qb,
                                              const bf16_t* __restrict__ Kb,
                                              const bf16_t* __restrict__ Vt,
                                              bf16_t* __restrict__ Cx) {
  __shared__ float  sS[4][32][33];
  __shared__ __align__(32) bf16_t sP[4][32][32];
  __shared__ float  sCorr[4][32];
  __shared__ float  sSum[4][32];

  const int lane = threadIdx.x & 31;
  const int wv   = threadIdx.x >> 5;
  const int r    = lane & 15;
  const int bsel = lane >> 4;
  const int head = blockIdx.y;
  const int q0   = blockIdx.x * 128 + wv * 32;

  const bf16_t* Qh = Qb + (size_t)head * S_LEN * HSZ;
  const bf16_t* Kh = Kb + (size_t)head * S_LEN * HSZ;
  const bf16_t* Vh = Vt + (size_t)head * HSZ * S_LEN;

  // Q fragments held in registers: 2 q-tiles x 4 d-chunks
  v16bf qf[2][4];
#pragma unroll
  for (int t = 0; t < 2; ++t)
#pragma unroll
    for (int c = 0; c < 4; ++c)
      qf[t][c] = load_a_frag(Qh + (size_t)(q0 + 16 * t + r) * HSZ + 32 * c + 16 * bsel);

  v8f acc[2][8];
#pragma unroll
  for (int t = 0; t < 2; ++t)
#pragma unroll
    for (int dt = 0; dt < 8; ++dt) acc[t][dt] = vzero8();

  float m_run = -3.0e38f, l_run = 0.0f;
  const int   q_row = q0 + lane;
  const float scale = 0.088388347648318447f;   // 1/sqrt(128)
  const float log2e = 1.4426950408889634f;

  for (int kb0 = 0; kb0 < q0 + 32; kb0 += 32) {   // causal block skip
    // ---- scores = Q K^T -------------------------------------------------
    v16bf kf[2][4];
#pragma unroll
    for (int kt = 0; kt < 2; ++kt)
#pragma unroll
      for (int c = 0; c < 4; ++c)
        kf[kt][c] = *(const v16bf*)(Kh + (size_t)(kb0 + 16 * kt + r) * HSZ + 32 * c + 16 * bsel);

    v8f sc[2][2];
#pragma unroll
    for (int qt = 0; qt < 2; ++qt)
#pragma unroll
      for (int kt = 0; kt < 2; ++kt) {
        sc[qt][kt] = vzero8();
#pragma unroll
        for (int c = 0; c < 4; ++c)
          sc[qt][kt] = wmma_bf16(qf[qt][c], kf[kt][c], sc[qt][kt]);
      }

#pragma unroll
    for (int qt = 0; qt < 2; ++qt)
#pragma unroll
      for (int kt = 0; kt < 2; ++kt)
#pragma unroll
        for (int v = 0; v < 8; ++v)
          sS[wv][16 * qt + v + 8 * bsel][16 * kt + r] = sc[qt][kt][v];

    // ---- online softmax, lane = query row ------------------------------
    float mx = m_run;
#pragma unroll 8
    for (int j = 0; j < 32; ++j) {
      float s = sS[wv][lane][j] * scale;
      if (kb0 + j > q_row) s -= 65504.0f;        // reference additive mask
      mx = fmaxf(mx, s);
    }
    const float corr = exp2f((m_run - mx) * log2e);
    float lsum = 0.0f;
#pragma unroll 8
    for (int j = 0; j < 32; ++j) {
      float s = sS[wv][lane][j] * scale;
      if (kb0 + j > q_row) s -= 65504.0f;
      float p = exp2f((s - mx) * log2e);
      lsum += p;
      sP[wv][lane][j] = (bf16_t)p;
    }
    l_run = l_run * corr + lsum;
    m_run = mx;
    sCorr[wv][lane] = corr;

    // ---- rescale accumulators, build P A-frags, ctx += P V -------------
    v16bf pf[2];
#pragma unroll
    for (int qt = 0; qt < 2; ++qt) {
      const bf16_t* pp = &sP[wv][16 * qt + r][0] + 16 * bsel;
      v8bf lo = *(const v8bf*)pp;
      v8bf hi = *(const v8bf*)(pp + 8);
      pf[qt] = cat16(lo, hi);
      float cr[8];
#pragma unroll
      for (int v = 0; v < 8; ++v) cr[v] = sCorr[wv][16 * qt + v + 8 * bsel];
#pragma unroll
      for (int dt = 0; dt < 8; ++dt)
#pragma unroll
        for (int v = 0; v < 8; ++v) acc[qt][dt][v] *= cr[v];
    }
#pragma unroll
    for (int dt = 0; dt < 8; ++dt) {
      v16bf vf = *(const v16bf*)(Vh + (size_t)(16 * dt + r) * S_LEN + kb0 + 16 * bsel);
#pragma unroll
      for (int qt = 0; qt < 2; ++qt) acc[qt][dt] = wmma_bf16(pf[qt], vf, acc[qt][dt]);
    }
  }

  // ---- finalize: ctx = acc / l, write bf16 [s][head*128+d] --------------
  sSum[wv][lane] = l_run;
#pragma unroll
  for (int qt = 0; qt < 2; ++qt) {
    float inv[8];
#pragma unroll
    for (int v = 0; v < 8; ++v) inv[v] = 1.0f / sSum[wv][16 * qt + v + 8 * bsel];
#pragma unroll
    for (int dt = 0; dt < 8; ++dt) {
      const int d = 16 * dt + r;
#pragma unroll
      for (int v = 0; v < 8; ++v) {
        const int row = q0 + 16 * qt + v + 8 * bsel;
        Cx[(size_t)row * HID + head * HSZ + d] = (bf16_t)(acc[qt][dt][v] * inv[v]);
      }
    }
  }
}

// ---------------------------------------------------------------------------
// GEMM2: out = ctx @ W_dense + b_dense   (fp32 output), wave tile 64x64
// ---------------------------------------------------------------------------
__global__ __launch_bounds__(256) void k_out_gemm(const bf16_t* __restrict__ A,
                                                  const bf16_t* __restrict__ Wt,
                                                  const float* __restrict__ bias,
                                                  float* __restrict__ out) {
  const int lane = threadIdx.x & 31;
  const int w    = threadIdx.x >> 5;
  const int r    = lane & 15;
  const int bsel = lane >> 4;
  const int m_base = blockIdx.y * 128 + (w & 1) * 64;
  const int n_base = blockIdx.x * 256 + (w >> 1) * 64;

  v8f acc[4][4];
#pragma unroll
  for (int i = 0; i < 4; ++i)
#pragma unroll
    for (int j = 0; j < 4; ++j) acc[i][j] = vzero8();

  for (int k0 = 0; k0 < HID; k0 += 32) {
    v16bf af[4];
#pragma unroll
    for (int i = 0; i < 4; ++i)
      af[i] = load_a_frag(A + (size_t)(m_base + 16 * i + r) * HID + k0 + 16 * bsel);
    v16bf bfr[4];
#pragma unroll
    for (int j = 0; j < 4; ++j)
      bfr[j] = *(const v16bf*)(Wt + (size_t)(n_base + 16 * j + r) * HID + k0 + 16 * bsel);
#pragma unroll
    for (int i = 0; i < 4; ++i)
#pragma unroll
      for (int j = 0; j < 4; ++j) acc[i][j] = wmma_bf16(af[i], bfr[j], acc[i][j]);
  }

#pragma unroll
  for (int j = 0; j < 4; ++j) {
    const int n    = n_base + 16 * j + r;
    const float bv = bias[n];
#pragma unroll
    for (int i = 0; i < 4; ++i)
#pragma unroll
      for (int v = 0; v < 8; ++v) {
        const int row = m_base + 16 * i + v + 8 * bsel;
        out[(size_t)row * HID + n] = acc[i][j][v] + bv;
      }
  }
}

// ---------------------------------------------------------------------------
extern "C" void kernel_launch(void* const* d_in, const int* in_sizes, int n_in,
                              void* d_out, int out_size, void* d_ws, size_t ws_size,
                              hipStream_t stream) {
  const float* hidden = (const float*)d_in[0];
  // d_in[1] = ltor_mask: causality is applied analytically, not needed.
  const float* W_qkv   = (const float*)d_in[2];
  const float* b_qkv   = (const float*)d_in[3];
  const float* W_dense = (const float*)d_in[4];
  const float* b_dense = (const float*)d_in[5];
  float* out = (float*)d_out;

  char* ws = (char*)d_ws;                      // 72 MB total
  bf16_t* Xb  = (bf16_t*)(ws);                 //  8 MB  [S][H] bf16
  bf16_t* Wqt = (bf16_t*)(ws + ((size_t)8  << 20));  // 24 MB [6144][2048]
  bf16_t* Wdt = (bf16_t*)(ws + ((size_t)32 << 20));  //  8 MB [2048][2048]
  bf16_t* Qb  = (bf16_t*)(ws + ((size_t)40 << 20));  //  8 MB [NH][S][HS]
  bf16_t* Kb  = (bf16_t*)(ws + ((size_t)48 << 20));  //  8 MB [NH][S][HS]
  bf16_t* Vt  = (bf16_t*)(ws + ((size_t)56 << 20));  //  8 MB [NH][HS][S]
  bf16_t* Cx  = (bf16_t*)(ws + ((size_t)64 << 20));  //  8 MB [S][H]

  const int nX = S_LEN * HID;
  k_cast_bf16<<<nX / 256, 256, 0, stream>>>(hidden, Xb, nX);
  k_transpose_cast<<<dim3(NQKV / 32, HID / 32), dim3(32, 8), 0, stream>>>(W_qkv, Wqt, HID, NQKV);
  k_transpose_cast<<<dim3(HID / 32, HID / 32), dim3(32, 8), 0, stream>>>(W_dense, Wdt, HID, HID);

  k_qkv_gemm<<<dim3(NQKV / 256, S_LEN / 128), 256, 0, stream>>>(Xb, Wqt, b_qkv, Qb, Kb, Vt);
  k_attn<<<dim3(S_LEN / 128, NHEAD), 128, 0, stream>>>(Qb, Kb, Vt, Cx);
  k_out_gemm<<<dim3(HID / 256, S_LEN / 128), 256, 0, stream>>>(Cx, Wdt, b_dense, out);
}
